// LSTMAttention_10290741641310
// MI455X (gfx1250) — matrix operand order repaired and checked by reading, not compiled, not run on hardware
//
#include <hip/hip_runtime.h>
#include <hip/hip_bf16.h>
#include <math.h>

// Problem dims (fixed by reference)
#define Bc   64
#define Tc   256
#define Sc   512
#define Ec   512
#define Hc   1024
#define G4H  4096   // 4*H

typedef __attribute__((ext_vector_type(16))) __bf16 v16bf;
typedef __attribute__((ext_vector_type(8)))  __bf16 v8bf;
typedef __attribute__((ext_vector_type(8)))  float  v8f;

__device__ __forceinline__ v8f zero8() {
  v8f z;
#pragma unroll
  for (int i = 0; i < 8; ++i) z[i] = 0.0f;
  return z;
}

__device__ __forceinline__ v8f wmma_bf16(v16bf a, v16bf b, v8f c) {
  // D = A(16x32 bf16) * B(32x16 bf16) + C(16x16 f32)
  return __builtin_amdgcn_wmma_f32_16x16x32_bf16(
      /*neg_a=*/false, a, /*neg_b=*/false, b,
      /*c_mod=*/(short)0, c, /*reuse_a=*/false, /*reuse_b=*/false);
}

// A fragment: 16x32 (MxK) bf16, row-major source with leading dim lda.
// Lane l (l<16): row m=l, holds K 0..7 (v0..v3) and 16..23 (v4..v7).
// Lane l (l>=16): row m=l-16, holds K 8..15 and 24..31.
__device__ __forceinline__ v16bf load_a_frag(const __bf16* A, size_t lda,
                                             int row0, int kk, int lane) {
  int m = lane & 15, g = lane >> 4;
  const __bf16* p = A + (size_t)(row0 + m) * lda + (size_t)(kk + g * 8);
  v8bf lo = *(const v8bf*)p;          // 16B contiguous
  v8bf hi = *(const v8bf*)(p + 16);   // +32B
  v16bf r;
#pragma unroll
  for (int i = 0; i < 8; ++i) { r[i] = lo[i]; r[i + 8] = hi[i]; }
  return r;
}

// B fragment: 32x16 (KxN) bf16 from a PRE-TRANSPOSED weight [N][K] (ldb=K).
// Lane l: col n=l&15, group g=l>>4 holds K kk+g*8..+7 and kk+16+g*8..+7.
__device__ __forceinline__ v16bf load_b_frag(const __bf16* Bt, size_t ldb,
                                             int col0, int kk, int lane) {
  int n = lane & 15, g = lane >> 4;
  const __bf16* p = Bt + (size_t)(col0 + n) * ldb + (size_t)(kk + g * 8);
  v8bf lo = *(const v8bf*)p;
  v8bf hi = *(const v8bf*)(p + 16);
  v16bf r;
#pragma unroll
  for (int i = 0; i < 8; ++i) { r[i] = lo[i]; r[i + 8] = hi[i]; }
  return r;
}

// One 32-deep K step of a 32x64 output tile: 2 A-frags x 4 B-frags -> 8 WMMA.
// Each B fragment reused by 2 WMMAs, each A fragment by 4.
__device__ __forceinline__ void mma_step_32x64(const __bf16* A, size_t lda,
                                               const __bf16* Bt, size_t ldb,
                                               int row0, int col0, int kk,
                                               int lane, v8f acc[2][4]) {
  v16bf a0 = load_a_frag(A, lda, row0,      kk, lane);
  v16bf a1 = load_a_frag(A, lda, row0 + 16, kk, lane);
#pragma unroll
  for (int j = 0; j < 4; ++j) {
    v16bf b = load_b_frag(Bt, ldb, col0 + 16 * j, kk, lane);
    acc[0][j] = wmma_bf16(a0, b, acc[0][j]);
    acc[1][j] = wmma_bf16(a1, b, acc[1][j]);
  }
}

__device__ __forceinline__ float sigmoidf(float x) {
  return 1.0f / (1.0f + __expf(-x));
}

// ---------------- prep kernels ----------------
__global__ void pack_bf16_kernel(const float* __restrict__ s,
                                 __bf16* __restrict__ d, int n) {
  int i = blockIdx.x * 256 + threadIdx.x;
  if (i < n) d[i] = (__bf16)s[i];
}

// src [K][N] fp32 (row-major)  ->  dst [N][K] bf16
__global__ void transpose_bf16_kernel(const float* __restrict__ s,
                                      __bf16* __restrict__ d, int K, int N) {
  int i = blockIdx.x * 256 + threadIdx.x;
  if (i >= K * N) return;
  int k = i / N, n = i - k * N;
  d[(size_t)n * K + k] = (__bf16)s[i];
}

// ---------------- per-step kernels ----------------
// gates[64][4096] = x_t @ W_ih + h @ W_hh + b_ih + b_hh  (bf16 WMMA, f32 acc)
// 32x64 tile per wave: 2 row-blocks x 64 col-tiles = 128 waves = 16 blocks.
__global__ void gemm_gates_kernel(const __bf16* __restrict__ xbf,
                                  const __bf16* __restrict__ WihT,
                                  const __bf16* __restrict__ hbf,
                                  const __bf16* __restrict__ WhhT,
                                  const float* __restrict__ b_ih,
                                  const float* __restrict__ b_hh,
                                  float* __restrict__ gates, int t) {
  int wid  = (int)((blockIdx.x * blockDim.x + threadIdx.x) >> 5);
  int lane = threadIdx.x & 31;
  const int NT = G4H / 64;
  if (wid >= (Bc / 32) * NT) return;
  int row0 = (wid / NT) * 32;
  int col0 = (wid % NT) * 64;
  v8f acc[2][4];
#pragma unroll
  for (int i = 0; i < 2; ++i)
#pragma unroll
    for (int j = 0; j < 4; ++j) acc[i][j] = zero8();

  // x projection: A = x_bf[b][t][:], lda = T*E, base offset t*E
  const __bf16* Ax = xbf + (size_t)t * Ec;
#pragma unroll 2
  for (int kk = 0; kk < Ec; kk += 32)
    mma_step_32x64(Ax, (size_t)Tc * Ec, WihT, Ec, row0, col0, kk, lane, acc);

  // h projection
#pragma unroll 2
  for (int kk = 0; kk < Hc; kk += 32) {
    if (kk + 32 < Hc)  // keep next weight k-slab warm (global_prefetch_b8)
      __builtin_prefetch(
          (const void*)(WhhT + (size_t)(col0 + (lane & 15)) * Hc + kk + 32), 0, 1);
    mma_step_32x64(hbf, Hc, WhhT, Hc, row0, col0, kk, lane, acc);
  }

  int n = lane & 15, g = lane >> 4;
#pragma unroll
  for (int i = 0; i < 2; ++i)
#pragma unroll
    for (int j = 0; j < 4; ++j) {
      int ccol = col0 + 16 * j + n;
      float bias = b_ih[ccol] + b_hh[ccol];
#pragma unroll
      for (int q = 0; q < 8; ++q) {
        int r = row0 + 16 * i + q + 8 * g;   // C layout: vgpr q -> row q (+8 hi lanes)
        gates[(size_t)r * G4H + ccol] = acc[i][j][q] + bias;
      }
    }
}

// LSTM cell: gates -> c_new, hy; hy stored as bf16 into cat[:,H:2H]
__global__ void lstm_cell_kernel(const float* __restrict__ gates,
                                 float* __restrict__ c,
                                 __bf16* __restrict__ catbf,
                                 float* __restrict__ cT) {
  int idx = blockIdx.x * 256 + threadIdx.x;
  if (idx >= Bc * Hc) return;
  int b = idx >> 10, h = idx & (Hc - 1);
  const float* g = gates + (size_t)b * G4H;
  float i  = sigmoidf(g[h]);
  float f  = sigmoidf(g[Hc + h]);
  float gg = tanhf(g[2 * Hc + h]);
  float o  = sigmoidf(g[3 * Hc + h]);
  float cn = f * c[idx] + i * gg;
  c[idx] = cn;
  float hy = o * tanhf(cn);
  catbf[(size_t)b * (2 * Hc) + Hc + h] = (__bf16)hy;
  if (cT) cT[idx] = cn;   // final cell state at t==T-1
}

// target[64][1024] = hy @ W_in  (A = cat[:,H:2H], lda=2H); 32 waves = 4 blocks
__global__ void gemm_target_kernel(const __bf16* __restrict__ catbf,
                                   const __bf16* __restrict__ WinT,
                                   float* __restrict__ target) {
  int wid  = (int)((blockIdx.x * blockDim.x + threadIdx.x) >> 5);
  int lane = threadIdx.x & 31;
  const int NT = Hc / 64;
  if (wid >= (Bc / 32) * NT) return;
  int row0 = (wid / NT) * 32;
  int col0 = (wid % NT) * 64;
  v8f acc[2][4];
#pragma unroll
  for (int i = 0; i < 2; ++i)
#pragma unroll
    for (int j = 0; j < 4; ++j) acc[i][j] = zero8();

  const __bf16* A = catbf + Hc;          // hy half
#pragma unroll 2
  for (int kk = 0; kk < Hc; kk += 32)
    mma_step_32x64(A, 2 * Hc, WinT, Hc, row0, col0, kk, lane, acc);

  int n = lane & 15, g = lane >> 4;
#pragma unroll
  for (int i = 0; i < 2; ++i)
#pragma unroll
    for (int j = 0; j < 4; ++j)
#pragma unroll
      for (int q = 0; q < 8; ++q)
        target[(size_t)(row0 + 16 * i + q + 8 * g) * Hc + col0 + 16 * j + n] =
            acc[i][j][q];
}

// Per-batch attention: scores = ctx_b . target_b, softmax, wctx -> cat[:,0:H]
__global__ void attn_kernel(const float* __restrict__ ctx,
                            const float* __restrict__ target,
                            __bf16* __restrict__ catbf) {
  __shared__ float sc[Sc];
  __shared__ float red[8];
  int b = blockIdx.x;
  int lane = threadIdx.x & 31, wv = threadIdx.x >> 5;
  const float* ctxb = ctx + (size_t)b * Sc * Hc;
  const float* tb   = target + (size_t)b * Hc;

  // scores: one wave per s (coalesced 128B rows, wave32 xor-reduce)
  for (int s = wv; s < Sc; s += 8) {
    const float* row = ctxb + (size_t)s * Hc;
    float acc = 0.0f;
    for (int k = lane; k < Hc; k += 32) acc = fmaf(row[k], tb[k], acc);
#pragma unroll
    for (int off = 16; off > 0; off >>= 1) acc += __shfl_xor(acc, off, 32);
    if (lane == 0) sc[s] = acc;
  }
  __syncthreads();

  // softmax (max then sum), block-wide
  float m = -3.0e38f;
  for (int s = threadIdx.x; s < Sc; s += 256) m = fmaxf(m, sc[s]);
#pragma unroll
  for (int off = 16; off > 0; off >>= 1) m = fmaxf(m, __shfl_xor(m, off, 32));
  if (lane == 0) red[wv] = m;
  __syncthreads();
  m = red[0];
#pragma unroll
  for (int w = 1; w < 8; ++w) m = fmaxf(m, red[w]);
  __syncthreads();

  float psum = 0.0f;
  for (int s = threadIdx.x; s < Sc; s += 256) {
    float e = __expf(sc[s] - m);
    sc[s] = e;
    psum += e;
  }
#pragma unroll
  for (int off = 16; off > 0; off >>= 1) psum += __shfl_xor(psum, off, 32);
  if (lane == 0) red[wv] = psum;
  __syncthreads();
  float sum = 0.0f;
#pragma unroll
  for (int w = 0; w < 8; ++w) sum += red[w];
  float inv = 1.0f / sum;

  // wctx[h] = sum_s attn[s] * ctx[b][s][h]  (coalesced over h)
  for (int h = threadIdx.x; h < Hc; h += 256) {
    float acc = 0.0f;
    for (int s = 0; s < Sc; ++s) acc = fmaf(sc[s], ctxb[(size_t)s * Hc + h], acc);
    catbf[(size_t)b * (2 * Hc) + h] = (__bf16)(acc * inv);
  }
}

// h_tilde = tanh(cat @ W_out); writes out[b][t][:], feeds h_bf, optional hT
__global__ void gemm_out_kernel(const __bf16* __restrict__ catbf,
                                const __bf16* __restrict__ WoutT,
                                float* __restrict__ out,
                                __bf16* __restrict__ hbf,
                                float* __restrict__ hT, int t) {
  int wid  = (int)((blockIdx.x * blockDim.x + threadIdx.x) >> 5);
  int lane = threadIdx.x & 31;
  const int NT = Hc / 64;
  if (wid >= (Bc / 32) * NT) return;     // 32 waves, K = 2048
  int row0 = (wid / NT) * 32;
  int col0 = (wid % NT) * 64;
  v8f acc[2][4];
#pragma unroll
  for (int i = 0; i < 2; ++i)
#pragma unroll
    for (int j = 0; j < 4; ++j) acc[i][j] = zero8();

#pragma unroll 2
  for (int kk = 0; kk < 2 * Hc; kk += 32) {
    if (kk + 32 < 2 * Hc)
      __builtin_prefetch(
          (const void*)(WoutT + (size_t)(col0 + (lane & 15)) * (2 * Hc) + kk + 32), 0, 1);
    mma_step_32x64(catbf, 2 * Hc, WoutT, 2 * Hc, row0, col0, kk, lane, acc);
  }

  int n = lane & 15, g = lane >> 4;
#pragma unroll
  for (int i = 0; i < 2; ++i)
#pragma unroll
    for (int j = 0; j < 4; ++j)
#pragma unroll
      for (int q = 0; q < 8; ++q) {
        int r = row0 + 16 * i + q + 8 * g;      // batch index b
        int ccol = col0 + 16 * j + n;
        float v = tanhf(acc[i][j][q]);
        out[(size_t)r * Tc * Hc + (size_t)t * Hc + ccol] = v;
        hbf[(size_t)r * Hc + ccol] = (__bf16)v;
        if (hT) hT[(size_t)r * Hc + ccol] = v;
      }
}

// ---------------- host ----------------
extern "C" void kernel_launch(void* const* d_in, const int* in_sizes, int n_in,
                              void* d_out, int out_size, void* d_ws, size_t ws_size,
                              hipStream_t stream) {
  (void)in_sizes; (void)n_in; (void)out_size; (void)ws_size;
  const float* x     = (const float*)d_in[0];
  const float* h0    = (const float*)d_in[1];
  const float* c0    = (const float*)d_in[2];
  const float* ctx   = (const float*)d_in[3];
  const float* W_ih  = (const float*)d_in[4];
  const float* b_ih  = (const float*)d_in[5];
  const float* W_hh  = (const float*)d_in[6];
  const float* b_hh  = (const float*)d_in[7];
  const float* W_in  = (const float*)d_in[8];
  const float* W_out = (const float*)d_in[9];
  float* out = (float*)d_out;

  char* ws = (char*)d_ws;
  size_t off = 0;
  auto alloc = [&](size_t bytes) -> void* {
    off = (off + 255) & ~(size_t)255;
    void* p = ws + off;
    off += bytes;
    return p;
  };
  __bf16* x_bf   = (__bf16*)alloc((size_t)Bc * Tc * Ec * 2);      // 16 MB
  __bf16* WihT   = (__bf16*)alloc((size_t)G4H * Ec * 2);          // 4 MB
  __bf16* WhhT   = (__bf16*)alloc((size_t)G4H * Hc * 2);          // 8 MB
  __bf16* WinT   = (__bf16*)alloc((size_t)Hc * Hc * 2);           // 2 MB
  __bf16* WoutT  = (__bf16*)alloc((size_t)Hc * 2 * Hc * 2);       // 4 MB
  float*  gates  = (float*) alloc((size_t)Bc * G4H * 4);          // 1 MB
  float*  c_buf  = (float*) alloc((size_t)Bc * Hc * 4);
  __bf16* cat_bf = (__bf16*)alloc((size_t)Bc * 2 * Hc * 2);       // [wctx | hy]
  float*  target = (float*) alloc((size_t)Bc * Hc * 4);
  __bf16* h_bf   = (__bf16*)alloc((size_t)Bc * Hc * 2);

  int n;
  n = Bc * Tc * Ec;
  pack_bf16_kernel<<<(n + 255) / 256, 256, 0, stream>>>(x, x_bf, n);
  n = Bc * Hc;
  pack_bf16_kernel<<<(n + 255) / 256, 256, 0, stream>>>(h0, h_bf, n);
  hipMemcpyAsync(c_buf, c0, (size_t)Bc * Hc * 4, hipMemcpyDeviceToDevice, stream);
  n = Ec * G4H;
  transpose_bf16_kernel<<<(n + 255) / 256, 256, 0, stream>>>(W_ih, WihT, Ec, G4H);
  n = Hc * G4H;
  transpose_bf16_kernel<<<(n + 255) / 256, 256, 0, stream>>>(W_hh, WhhT, Hc, G4H);
  n = Hc * Hc;
  transpose_bf16_kernel<<<(n + 255) / 256, 256, 0, stream>>>(W_in, WinT, Hc, Hc);
  n = 2 * Hc * Hc;
  transpose_bf16_kernel<<<(n + 255) / 256, 256, 0, stream>>>(W_out, WoutT, 2 * Hc, Hc);

  float* hT = out + (size_t)Bc * Tc * Hc;
  float* cT = hT + (size_t)Bc * Hc;

  for (int t = 0; t < Tc; ++t) {
    bool last = (t == Tc - 1);
    // gates: 128 waves -> 16 blocks x 8 waves
    gemm_gates_kernel<<<16, 256, 0, stream>>>(x_bf, WihT, h_bf, WhhT,
                                              b_ih, b_hh, gates, t);
    lstm_cell_kernel<<<(Bc * Hc + 255) / 256, 256, 0, stream>>>(
        gates, c_buf, cat_bf, last ? cT : nullptr);
    gemm_target_kernel<<<4, 256, 0, stream>>>(cat_bf, WinT, target);
    attn_kernel<<<Bc, 256, 0, stream>>>(ctx, target, cat_bf);
    gemm_out_kernel<<<4, 256, 0, stream>>>(cat_bf, WoutT, out, h_bf,
                                           last ? hT : nullptr, t);
  }
}
